// CustomEncoder_47828755808303
// MI455X (gfx1250) — matrix operand order repaired and checked by reading, not compiled, not run on hardware
//
#include <hip/hip_runtime.h>

typedef __attribute__((ext_vector_type(16))) _Float16 v16h;
typedef __attribute__((ext_vector_type(8)))  _Float16 v8h;
typedef __attribute__((ext_vector_type(8)))  float    v8f;

#define BATCH 512
#define NPTS  64

// Pack 8 f32 -> 8 contiguous channels (NHWC): one 16B f16 store or 32B f32 store.
__device__ __forceinline__ void store8(_Float16* dst, const float* y) {
  v8h h;
#pragma unroll
  for (int r = 0; r < 8; ++r) h[r] = (_Float16)y[r];
  *(v8h*)dst = h;
}
__device__ __forceinline__ void store8(float* dst, const float* y) {
  v8f f;
#pragma unroll
  for (int r = 0; r < 8; ++r) f[r] = y[r];
  *(v8f*)dst = f;
}

// ---------------------------------------------------------------------------
// Layer 0: IC=1, K=9 (zero-padded to 32). Scalar im2col gather (tiny layer),
// NCHW f32 input -> NHWC f16 output [512,48,48,32].
// Per 256-thread block: oc-tile 16, n-tile 128 (16 pixels per wave).
// ---------------------------------------------------------------------------
__global__ __launch_bounds__(256)
void conv0_wmma(const float* __restrict__ x, const float* __restrict__ w,
                const float* __restrict__ bias, const float* __restrict__ gamma,
                const float* __restrict__ beta, const float* __restrict__ mean,
                const float* __restrict__ var, _Float16* __restrict__ out) {
  constexpr int TILES_N = BATCH * 48 * 48 / 128;  // 9216
  __shared__ _Float16 sA[16 * 32];
  __shared__ float sSc[16], sSh[16];

  const int tid     = threadIdx.x;
  const int tile_n  = blockIdx.x % TILES_N;
  const int oc_base = (blockIdx.x / TILES_N) * 16;

  for (int i = tid; i < 512; i += 256) {
    const int m = i >> 5, k = i & 31;
    sA[i] = (k < 9) ? (_Float16)w[(oc_base + m) * 9 + k] : (_Float16)0.0f;
  }
  if (tid < 16) {
    const int oc = oc_base + tid;
    const float sc = gamma[oc] * rsqrtf(var[oc] + 1e-5f);
    sSc[tid] = sc;
    sSh[tid] = (bias[oc] - mean[oc]) * sc + beta[oc];
  }
  __syncthreads();

  const int wave = tid >> 5, lane = tid & 31;
  const int mrow = lane & 15;
  const int khA  = (lane >> 4) * 16;
  const int khB  = (lane >> 4) * 16;

  const int n   = tile_n * 128 + wave * 16 + (lane & 15);
  const int b   = n / 2304;
  const int rem = n - b * 2304;
  const int oy  = rem / 48, ox = rem - oy * 48;
  const int iy0 = 2 * oy - 1, ix0 = 2 * ox - 1;
  const float* xb = x + (long)b * 9216;

  v16h af, bf;
#pragma unroll
  for (int e = 0; e < 8; ++e) {               // A: lane holds k = khA+e (contiguous 16)
    af[e]     = sA[mrow * 32 + khA + e];
    af[8 + e] = sA[mrow * 32 + khA + 8 + e];
  }
#pragma unroll
  for (int e = 0; e < 16; ++e) {              // B: lane holds k = khB+e
    const int k = khB + e;
    float val = 0.0f;
    if (k < 9) {
      const int ky = k / 3, kx = k - (k / 3) * 3;
      const int iy = iy0 + ky, ix = ix0 + kx;
      if ((unsigned)iy < 96u && (unsigned)ix < 96u) val = xb[iy * 96 + ix];
    }
    bf[e] = (_Float16)val;
  }

  v8f acc = {};
  acc = __builtin_amdgcn_wmma_f32_16x16x32_f16(false, af, false, bf,
                                               (short)0, acc, false, false);

  const int oc0 = (lane >> 4) * 8;
  float y[8];
#pragma unroll
  for (int r = 0; r < 8; ++r) {
    float t = acc[r] * sSc[oc0 + r] + sSh[oc0 + r];
    y[r] = (t >= 0.0f) ? t : 0.01f * t;
  }
  store8(out + (((long)b * 48 + oy) * 48 + ox) * 32 + oc_base + oc0, y);
}

// ---------------------------------------------------------------------------
// Layers 1-3: NHWC f16 in, NHWC out. K reordered as k = (ky*3+kx)*IC + ic so a
// lane's B fragment is 16 contiguous channels == one aligned 32B vector load
// (A and B both use contiguous k = 16*h + s). Per wave: M=16 x N=64 (4 accums),
// A fragment reused across 4 WMMAs. Block n-tile = 512 pixels.
// ---------------------------------------------------------------------------
template<typename TOUT, int IC, int OC, int IH, int IW, int OH, int OW>
__global__ __launch_bounds__(256)
void convn_wmma(const _Float16* __restrict__ in, const float* __restrict__ w,
                const float* __restrict__ bias, const float* __restrict__ gamma,
                const float* __restrict__ beta, const float* __restrict__ mean,
                const float* __restrict__ var, TOUT* __restrict__ out) {
  constexpr int K   = IC * 9;            // multiple of 32 (IC in {32,64,128})
  constexpr int OHW = OH * OW;
  constexpr int NT  = BATCH * OHW;
  constexpr int TILES_N = NT / 512;

  __shared__ _Float16 sA[16 * K];
  __shared__ float sSc[16], sSh[16];

  const int tid     = threadIdx.x;
  const int tile_n  = blockIdx.x % TILES_N;
  const int oc_base = (blockIdx.x / TILES_N) * 16;

  // Stage permuted weight panel: sA[m][kpos*IC+ic] = w[oc, ic, kpos]
  for (int i = tid; i < 16 * K; i += 256) {
    const int m = i / K, k = i - m * K;
    const int kpos = k / IC, ic = k - kpos * IC;   // IC is power of two
    sA[i] = (_Float16)w[((oc_base + m) * IC + ic) * 9 + kpos];
  }
  if (tid < 16) {
    const int oc = oc_base + tid;
    const float sc = gamma[oc] * rsqrtf(var[oc] + 1e-5f);
    sSc[tid] = sc;
    sSh[tid] = (bias[oc] - mean[oc]) * sc + beta[oc];
  }
  __syncthreads();

  const int wave = tid >> 5, lane = tid & 31;
  const int mrow = lane & 15;
  const int khA  = (lane >> 4) * 16;    // A: lanes 16-31 hold K+16 sub-run
  const int khB  = (lane >> 4) * 16;    // B: lanes 16-31 hold K+16 sub-run

  int bj[4], oyj[4], oxj[4];
  long ibase[4];
#pragma unroll
  for (int j = 0; j < 4; ++j) {
    const int n   = tile_n * 512 + wave * 64 + j * 16 + (lane & 15);
    const int b   = n / OHW;
    const int rem = n - b * OHW;
    bj[j]  = b;
    oyj[j] = rem / OW;
    oxj[j] = rem - oyj[j] * OW;
    ibase[j] = (long)b * IH * IW;
    __builtin_prefetch(in + ibase[j] * IC, 0, 3);  // gfx1250 global_prefetch
  }

  v8f acc[4] = {};
  for (int ky = 0; ky < 3; ++ky) {
    for (int kx = 0; kx < 3; ++kx) {
      const int kpos = ky * 3 + kx;
#pragma unroll
      for (int ics = 0; ics < IC / 32; ++ics) {
        const int kbase = kpos * IC + ics * 32;
        // A fragment: two aligned 16B LDS reads (16 contiguous k)
        const _Float16* ap = &sA[mrow * K + kbase + khA];
        const v8h a0 = *(const v8h*)ap;
        const v8h a1 = *(const v8h*)(ap + 8);
        v16h af;
#pragma unroll
        for (int e = 0; e < 8; ++e) { af[e] = a0[e]; af[8 + e] = a1[e]; }

        const int icb = ics * 32 + khB;
#pragma unroll
        for (int j = 0; j < 4; ++j) {
          const int iy = 2 * oyj[j] - 1 + ky;
          const int ix = 2 * oxj[j] - 1 + kx;
          v16h bf = {};
          if ((unsigned)iy < (unsigned)IH && (unsigned)ix < (unsigned)IW) {
            const _Float16* src = in + (ibase[j] + (long)iy * IW + ix) * IC + icb;
            bf = *(const v16h*)src;      // 32B aligned contiguous channels
          }
          acc[j] = __builtin_amdgcn_wmma_f32_16x16x32_f16(
              false, af, false, bf, (short)0, acc[j], false, false);
        }
      }
    }
  }

  // Epilogue: lane owns 8 consecutive oc -> contiguous in NHWC -> 1 store.
  const int oc0 = (lane >> 4) * 8;
#pragma unroll
  for (int j = 0; j < 4; ++j) {
    float y[8];
#pragma unroll
    for (int r = 0; r < 8; ++r) {
      float t = acc[j][r] * sSc[oc0 + r] + sSh[oc0 + r];
      y[r] = (t >= 0.0f) ? t : 0.01f * t;
    }
    TOUT* dst = out + (((long)bj[j] * OH + oyj[j]) * OW + oxj[j]) * OC + oc_base + oc0;
    store8(dst, y);
  }
}

// ---------------------------------------------------------------------------
// Point gather from NHWC f32 feature map [512,6,6,256]:
// out[b][c][j] = feat[b][yi][xi][c], xi = clip(trunc(px*5/95),0,5).
// ---------------------------------------------------------------------------
__global__ __launch_bounds__(256)
void gather_points(const float* __restrict__ feat, const float* __restrict__ p,
                   float* __restrict__ out) {
  const int t = blockIdx.x * 256 + threadIdx.x;   // exact: 512*256*64
  const int j = t & 63;
  const int c = (t >> 6) & 255;
  const int b = t >> 14;
  const float px = p[((b * NPTS) + j) * 2 + 0];
  const float py = p[((b * NPTS) + j) * 2 + 1];
  int xi = (int)(px * (5.0f / 95.0f));
  int yi = (int)(py * (5.0f / 95.0f));
  xi = xi < 0 ? 0 : (xi > 5 ? 5 : xi);
  yi = yi < 0 ? 0 : (yi > 5 ? 5 : yi);
  out[t] = feat[(((long)b * 6 + yi) * 6 + xi) * 256 + c];
}

// ---------------------------------------------------------------------------
// Workspace aliasing (NHWC activations):
//   region0 @0        : act0 f16 512*48*48*32 = 75,497,472 B ; reused by act2
//   region1 @75497472 : act1 f16 512*24*24*64 = 37,748,736 B ; reused by act3 f32
// total required ws = 113,246,208 B.
// ---------------------------------------------------------------------------
extern "C" void kernel_launch(void* const* d_in, const int* in_sizes, int n_in,
                              void* d_out, int out_size, void* d_ws, size_t ws_size,
                              hipStream_t stream) {
  (void)in_sizes; (void)n_in; (void)out_size; (void)ws_size;

  const float* x = (const float*)d_in[0];
  const float* p = (const float*)d_in[1];
  const float* W[4]; const float* Bb[4]; const float* G[4];
  const float* Be[4]; const float* Mu[4]; const float* Va[4];
  for (int li = 0; li < 4; ++li) {
    W[li]  = (const float*)d_in[2 + 6 * li + 0];
    Bb[li] = (const float*)d_in[2 + 6 * li + 1];
    G[li]  = (const float*)d_in[2 + 6 * li + 2];
    Be[li] = (const float*)d_in[2 + 6 * li + 3];
    Mu[li] = (const float*)d_in[2 + 6 * li + 4];
    Va[li] = (const float*)d_in[2 + 6 * li + 5];
  }

  char* ws = (char*)d_ws;
  _Float16* act0 = (_Float16*)ws;               // NHWC [512,48,48,32]
  _Float16* act1 = (_Float16*)(ws + 75497472);  // NHWC [512,24,24,64]
  _Float16* act2 = (_Float16*)ws;               // NHWC [512,12,12,128]
  float*    act3 = (float*)(ws + 75497472);     // NHWC [512,6,6,256] f32
  float* out = (float*)d_out;

  conv0_wmma<<<dim3((32 / 16) * (BATCH * 48 * 48 / 128)), dim3(256), 0, stream>>>(
      x, W[0], Bb[0], G[0], Be[0], Mu[0], Va[0], act0);

  convn_wmma<_Float16, 32, 64, 48, 48, 24, 24>
      <<<dim3((64 / 16) * (BATCH * 24 * 24 / 512)), dim3(256), 0, stream>>>(
          act0, W[1], Bb[1], G[1], Be[1], Mu[1], Va[1], act1);

  convn_wmma<_Float16, 64, 128, 24, 24, 12, 12>
      <<<dim3((128 / 16) * (BATCH * 12 * 12 / 512)), dim3(256), 0, stream>>>(
          act1, W[2], Bb[2], G[2], Be[2], Mu[2], Va[2], act2);

  convn_wmma<float, 128, 256, 12, 12, 6, 6>
      <<<dim3((256 / 16) * (BATCH * 6 * 6 / 512)), dim3(256), 0, stream>>>(
          act2, W[3], Bb[3], G[3], Be[3], Mu[3], Va[3], act3);

  gather_points<<<dim3(BATCH * 256 * NPTS / 256), dim3(256), 0, stream>>>(act3, p, out);
}
